// BetaVAEMark7Decoder_34376918237629
// MI455X (gfx1250) — matrix operand order repaired and checked by reading, not compiled, not run on hardware
//
#include <hip/hip_runtime.h>
#include <hip/hip_bf16.h>
#include <hip/hip_fp16.h>

typedef __attribute__((ext_vector_type(16))) _Float16 v16h;
typedef __attribute__((ext_vector_type(8)))  _Float16 v8h;
typedef __attribute__((ext_vector_type(4)))  _Float16 v4h;
typedef __attribute__((ext_vector_type(8)))  float    v8f;

union AFrag { v16h v; v8h h[2]; };

__device__ __forceinline__ float lrelu(float x) { return x > 0.f ? x : 0.01f * x; }
__device__ __forceinline__ float4 f4load(const float* p) { return *(const float4*)p; }
__device__ __forceinline__ void fma4(float4& a, float x, const float4 w) {
    a.x = fmaf(x, w.x, a.x); a.y = fmaf(x, w.y, a.y);
    a.z = fmaf(x, w.z, a.z); a.w = fmaf(x, w.w, a.w);
}
__device__ __forceinline__ v8h pack8(const float4& a0, const float4& a1) {
    v8h o;
    o[0] = (_Float16)a0.x; o[1] = (_Float16)a0.y; o[2] = (_Float16)a0.z; o[3] = (_Float16)a0.w;
    o[4] = (_Float16)a1.x; o[5] = (_Float16)a1.y; o[6] = (_Float16)a1.z; o[7] = (_Float16)a1.w;
    return o;
}
// load B fragment (one chunk) for this lane from fragment-ordered LDS
__device__ __forceinline__ v16h bfrag(const _Float16* s_bf, int c, int lane) {
    AFrag t;
    const _Float16* p = s_bf + (c * 32 + lane) * 16;
    t.h[0] = *(const v8h*)p;
    t.h[1] = *(const v8h*)(p + 8);
    return t.v;
}

// ---------------------------------------------------------------------------
// K1: linear(7->256)+lrelu -> NHWC (1,8,32); up1 k(5,2) s(2,2) -> (5,16,32)
//     tc1 3x3 s1 p1 (32->16) via WMMA + lrelu -> s2 (B,5,16,16) f16
// ---------------------------------------------------------------------------
__global__ __launch_bounds__(256) void k1_lin_up1_tc1(
    const float* __restrict__ latent, const float* __restrict__ w_lin,
    const float* __restrict__ b_lin,  const float* __restrict__ w_up1,
    const float* __restrict__ b_up1,  const float* __restrict__ w_tc1,
    const float* __restrict__ b_tc1,  _Float16* __restrict__ s2_out)
{
    __shared__ float    s_lat[7];
    __shared__ _Float16 s_x0[8 * 32];          // (1,8,32)
    __shared__ _Float16 s_s1[7 * 18 * 32];     // padded (5,16,32) + halo, 8064 B
    __shared__ _Float16 s_st[8 * 256];         // per-wave epilogue staging
    __shared__ _Float16 s_bf[9 * 32 * 16];     // tc1 B fragments, 9216 B

    const int b = blockIdx.x, tid = threadIdx.x;

    // zero halo buffer (v8h stores)
    for (int i = tid; i < (7 * 18 * 32) / 8; i += 256) ((v8h*)s_s1)[i] = (v8h){};
    // stage tc1 weight fragments once per block: [c][lane][i], B K-map = hi*16+i
    for (int idx = tid; idx < 9 * 32 * 16; idx += 256) {
        int c = idx >> 9, lane = (idx >> 4) & 31, i = idx & 15;
        int hi = lane >> 4, ln = lane & 15;
        s_bf[idx] = (_Float16)w_tc1[(c * 32 + hi * 16 + i) * 16 + ln];
    }
    if (tid < 7) s_lat[tid] = latent[b * 7 + tid];
    __syncthreads();

    // linear + lrelu; h[j], j==tid; x0[w][c] = h[c*8+w]
    {
        float acc = b_lin[tid];
#pragma unroll
        for (int i = 0; i < 7; ++i) acc = fmaf(s_lat[i], w_lin[i * 256 + tid], acc);
        s_x0[(tid & 7) * 32 + (tid >> 3)] = (_Float16)lrelu(acc);
    }
    __syncthreads();

    // up1: single tap per output (kh=oh, kw=ow&1, iw=ow>>1); 80 px * 4 oc-groups
    for (int t = tid; t < 320; t += 256) {
        int og = t & 3, p = t >> 2;
        int oh = p >> 4, ow = p & 15;
        int kw = ow & 1, iw = ow >> 1;
        const float* wbase = w_up1 + ((oh * 2 + kw) * 32) * 32 + og * 8;
        float4 a0 = f4load(b_up1 + og * 8), a1 = f4load(b_up1 + og * 8 + 4);
        const v8h* xp = (const v8h*)(s_x0 + iw * 32);
#pragma unroll
        for (int icg = 0; icg < 4; ++icg) {
            v8h xv = xp[icg];
#pragma unroll
            for (int j = 0; j < 8; ++j) {
                float xx = (float)xv[j];
                int ic = icg * 8 + j;
                fma4(a0, xx, f4load(wbase + ic * 32));
                fma4(a1, xx, f4load(wbase + ic * 32 + 4));
            }
        }
        *(v8h*)(s_s1 + ((oh + 1) * 18 + (ow + 1)) * 32 + og * 8) = pack8(a0, a1);
    }
    __syncthreads();

    // tc1: 80 px -> 5 tiles; K = 9 taps x 32ic -> 9 chunks
    const int wave = tid >> 5, lane = tid & 31;
    const int hi = lane >> 4, ln = lane & 15;
    if (wave < 5) {
        const int p0 = wave * 16;
        const int m = p0 + ln;
        const int oh = m >> 4, ow = m & 15;
        v8f acc = {};
#pragma unroll
        for (int c = 0; c < 9; ++c) {
            int kh = c / 3, kw = c % 3;
            const _Float16* sp = s_s1 + ((oh + 2 - kh) * 18 + (ow + 2 - kw)) * 32;
            AFrag A;
            A.h[0] = *(const v8h*)(sp + hi * 16);       // K = hi*16 + i
            A.h[1] = *(const v8h*)(sp + hi * 16 + 8);   // K = hi*16 + 8 + i
            acc = __builtin_amdgcn_wmma_f32_16x16x32_f16(false, A.v, false,
                                                         bfrag(s_bf, c, lane),
                                                         (short)0, acc, false, false);
        }
        float bias = b_tc1[ln];
        _Float16* st = s_st + wave * 256;
#pragma unroll
        for (int r = 0; r < 8; ++r)                      // D: M = r + 8*hi, N = ln
            st[(r + 8 * hi) * 16 + ln] = (_Float16)lrelu(acc[r] + bias);
        // same-wave DS ordering; coalesced 16B-per-lane writeout
        v8h ov = *(const v8h*)(st + lane * 8);
        *(v8h*)(s2_out + ((size_t)b * 80 + p0) * 16 + lane * 8) = ov;
    }
}

// ---------------------------------------------------------------------------
// K2: up2 k(5,2) s(5,2): (5,16,16)->(25,32,16); tc2 3x3 s1 p1 (16->8) WMMA
//     + lrelu -> s4 (B,25,32,8) f16
// ---------------------------------------------------------------------------
__global__ __launch_bounds__(256) void k2_up2_tc2(
    const _Float16* __restrict__ s2_in, const float* __restrict__ w_up2,
    const float* __restrict__ b_up2,    const float* __restrict__ w_tc2,
    const float* __restrict__ b_tc2,    _Float16* __restrict__ s4_out)
{
    __shared__ _Float16 s_s2[5 * 16 * 16];     // 2.5 KB
    __shared__ _Float16 s_s3[27 * 34 * 16];    // padded (25,32,16), 29376 B
    __shared__ _Float16 s_st[8 * 128];         // per-wave staging
    __shared__ _Float16 s_bf[5 * 32 * 16];     // tc2 B fragments, 5120 B

    const int b = blockIdx.x, tid = threadIdx.x;

    for (int i = tid; i < (27 * 34 * 16) / 8; i += 256) ((v8h*)s_s3)[i] = (v8h){};
    for (int i = tid; i < 160; i += 256)
        ((v8h*)s_s2)[i] = ((const v8h*)(s2_in + (size_t)b * 1280))[i];
    // tc2 B fragments: tap = 2c + (i>>3), ic = hi*8 + (i&7)
    for (int idx = tid; idx < 5 * 32 * 16; idx += 256) {
        int c = idx >> 9, lane = (idx >> 4) & 31, i = idx & 15;
        int hi = lane >> 4, ln = lane & 15;
        int tap = 2 * c + (i >> 3), ic = hi * 8 + (i & 7);
        _Float16 v = (_Float16)0.f;
        if (tap < 9 && ln < 8) v = (_Float16)w_tc2[(tap * 16 + ic) * 8 + ln];
        s_bf[idx] = v;
    }
    __syncthreads();

    // up2: kh=oh%5, ih=oh/5; kw=ow&1, iw=ow>>1; 800 px * 2 oc-groups
    for (int t = tid; t < 1600; t += 256) {
        int og = t & 1, p = t >> 1;
        int oh = p >> 5, ow = p & 31;
        int kh = oh % 5, ih = oh / 5;
        int kw = ow & 1, iw = ow >> 1;
        const float* wbase = w_up2 + ((kh * 2 + kw) * 16) * 16 + og * 8;
        float4 a0 = f4load(b_up2 + og * 8), a1 = f4load(b_up2 + og * 8 + 4);
        const v8h* xp = (const v8h*)(s_s2 + (ih * 16 + iw) * 16);
#pragma unroll
        for (int icg = 0; icg < 2; ++icg) {
            v8h xv = xp[icg];
#pragma unroll
            for (int j = 0; j < 8; ++j) {
                float xx = (float)xv[j];
                int ic = icg * 8 + j;
                fma4(a0, xx, f4load(wbase + ic * 16));
                fma4(a1, xx, f4load(wbase + ic * 16 + 4));
            }
        }
        *(v8h*)(s_s3 + ((oh + 1) * 34 + (ow + 1)) * 16 + og * 8) = pack8(a0, a1);
    }
    __syncthreads();

    // tc2: 800 px -> 50 tiles; K = 9 taps x 16ic -> 5 chunks (2 taps/chunk)
    const int wave = tid >> 5, lane = tid & 31;
    const int hi = lane >> 4, ln = lane & 15;
    const v8h zero8 = {};
    v16h Bf[5];
#pragma unroll
    for (int c = 0; c < 5; ++c) Bf[c] = bfrag(s_bf, c, lane);
    float bias = (ln < 8) ? b_tc2[ln] : 0.f;

    for (int t = wave; t < 50; t += 8) {
        int p0 = t * 16;
        int m = p0 + ln;
        int oh = m >> 5, ow = m & 31;
        v8f acc = {};
#pragma unroll
        for (int c = 0; c < 5; ++c) {
            AFrag A;
            { int tap = 2 * c, kh = tap / 3, kw = tap % 3;       // half0: ic = hi*8+i
              A.h[0] = *(const v8h*)(s_s3 + ((oh + 2 - kh) * 34 + (ow + 2 - kw)) * 16 + hi * 8); }
            if (2 * c + 1 < 9) {                                  // half1: tap 2c+1
              int tap = 2 * c + 1, kh = tap / 3, kw = tap % 3;
              A.h[1] = *(const v8h*)(s_s3 + ((oh + 2 - kh) * 34 + (ow + 2 - kw)) * 16 + hi * 8);
            } else A.h[1] = zero8;
            acc = __builtin_amdgcn_wmma_f32_16x16x32_f16(false, A.v, false, Bf[c],
                                                         (short)0, acc, false, false);
        }
        _Float16* st = s_st + wave * 128;
        if (ln < 8) {
#pragma unroll
            for (int r = 0; r < 8; ++r)
                st[(r + 8 * hi) * 8 + ln] = (_Float16)lrelu(acc[r] + bias);
        }
        v4h ov = *(const v4h*)(st + lane * 4);
        *(v4h*)(s4_out + ((size_t)b * 800 + p0) * 8 + lane * 4) = ov;
    }
}

// ---------------------------------------------------------------------------
// K3: up3 k(2,2) s(2,2): (25,32,8)->(50,64,8); tc3 3x3 s1 p1 (8->6) WMMA
//     + lrelu, coalesced NCHW f32 store
// ---------------------------------------------------------------------------
__global__ __launch_bounds__(256) void k3_up3_tc3(
    const _Float16* __restrict__ s4_in, const float* __restrict__ w_up3,
    const float* __restrict__ b_up3,    const float* __restrict__ w_tc3,
    const float* __restrict__ b_tc3,    float* __restrict__ out)
{
    __shared__ _Float16 s_s5[52 * 66 * 8];     // padded (50,64,8), 54912 B
    __shared__ float    s_st[8 * 96];          // per-wave 16px x 6ch staging, 3072 B
    __shared__ _Float16 s_bf[3 * 32 * 16];     // tc3 B fragments, 3072 B

    const int b = blockIdx.x, tid = threadIdx.x;

    for (int i = tid; i < (52 * 66 * 8) / 8; i += 256) ((v8h*)s_s5)[i] = (v8h){};
    // tc3 B fragments: K = hi*16+i -> tap = c*4 + 2*hi + (i>>3), ic = i&7
    for (int idx = tid; idx < 3 * 32 * 16; idx += 256) {
        int c = idx >> 9, lane = (idx >> 4) & 31, i = idx & 15;
        int hi = lane >> 4, ln = lane & 15;
        int tap = c * 4 + 2 * hi + (i >> 3), ic = i & 7;
        _Float16 v = (_Float16)0.f;
        if (tap < 9 && ln < 6) v = (_Float16)w_tc3[(tap * 8 + ic) * 6 + ln];
        s_bf[idx] = v;
    }
    __syncthreads();

    // up3: kh=oh&1, ih=oh>>1; kw=ow&1, iw=ow>>1; one thread = one px, 8 oc
    for (int p = tid; p < 3200; p += 256) {
        int oh = p >> 6, ow = p & 63;
        int kh = oh & 1, ih = oh >> 1;
        int kw = ow & 1, iw = ow >> 1;
        v8h xv = *(const v8h*)(s4_in + ((size_t)b * 800 + ih * 32 + iw) * 8);
        const float* wbase = w_up3 + ((kh * 2 + kw) * 8) * 8;
        float4 a0 = f4load(b_up3), a1 = f4load(b_up3 + 4);
#pragma unroll
        for (int ic = 0; ic < 8; ++ic) {
            float xx = (float)xv[ic];
            fma4(a0, xx, f4load(wbase + ic * 8));
            fma4(a1, xx, f4load(wbase + ic * 8 + 4));
        }
        *(v8h*)(s_s5 + ((oh + 1) * 66 + (ow + 1)) * 8) = pack8(a0, a1);
    }
    __syncthreads();

    // tc3: 3200 px -> 200 tiles; K = 9 taps x 8ic -> 3 chunks (4 taps/chunk)
    const int wave = tid >> 5, lane = tid & 31;
    const int hi = lane >> 4, ln = lane & 15;
    const v8h zero8 = {};
    v16h Bf[3];
#pragma unroll
    for (int c = 0; c < 3; ++c) Bf[c] = bfrag(s_bf, c, lane);
    float bias = (ln < 6) ? b_tc3[ln] : 0.f;

    for (int t = wave; t < 200; t += 8) {
        int p0 = t * 16;
        int m = p0 + ln;
        int oh = m >> 6, ow = m & 63;
        v8f acc = {};
#pragma unroll
        for (int c = 0; c < 3; ++c) {
            AFrag A;
            if (c == 2) {                       // taps 8..11: only tap 8 (hi==0 half)
                v8h v = *(const v8h*)(s_s5 + (oh * 66 + ow) * 8);  // kh=kw=2
                A.h[0] = hi ? zero8 : v;
                A.h[1] = zero8;
            } else {
                int tap0 = 4 * c + 2 * hi;      // half0: ic = i
                int kh0 = tap0 / 3, kw0 = tap0 % 3;
                A.h[0] = *(const v8h*)(s_s5 + ((oh + 2 - kh0) * 66 + (ow + 2 - kw0)) * 8);
                int tap1 = tap0 + 1;            // half1: ic = i-8
                int kh1 = tap1 / 3, kw1 = tap1 % 3;
                A.h[1] = *(const v8h*)(s_s5 + ((oh + 2 - kh1) * 66 + (ow + 2 - kw1)) * 8);
            }
            acc = __builtin_amdgcn_wmma_f32_16x16x32_f16(false, A.v, false, Bf[c],
                                                         (short)0, acc, false, false);
        }
        // stage tile (6 ch x 16 px f32) then write 64B-contiguous channel runs
        float* st = s_st + wave * 96;
        if (ln < 6) {
#pragma unroll
            for (int r = 0; r < 8; ++r)
                st[ln * 16 + (r + 8 * hi)] = lrelu(acc[r] + bias);
        }
        int oh_t = p0 >> 6, ow0 = p0 & 63;      // 16-px tile never crosses a row
#pragma unroll
        for (int j = 0; j < 3; ++j) {
            int val = j * 32 + lane;            // 0..95 = ch*16 + px
            int ch = val >> 4, px = val & 15;
            out[(((size_t)b * 6 + ch) * 50 + oh_t) * 64 + ow0 + px] = st[val];
        }
    }
}

// ---------------------------------------------------------------------------
extern "C" void kernel_launch(void* const* d_in, const int* in_sizes, int n_in,
                              void* d_out, int out_size, void* d_ws, size_t ws_size,
                              hipStream_t stream) {
    const float* latent = (const float*)d_in[0];
    const float* w_lin  = (const float*)d_in[1];
    const float* b_lin  = (const float*)d_in[2];
    const float* w_up1  = (const float*)d_in[3];
    const float* b_up1  = (const float*)d_in[4];
    const float* w_tc1  = (const float*)d_in[5];
    const float* b_tc1  = (const float*)d_in[6];
    const float* w_up2  = (const float*)d_in[7];
    const float* b_up2  = (const float*)d_in[8];
    const float* w_tc2  = (const float*)d_in[9];
    const float* b_tc2  = (const float*)d_in[10];
    const float* w_up3  = (const float*)d_in[11];
    const float* b_up3  = (const float*)d_in[12];
    const float* w_tc3  = (const float*)d_in[13];
    const float* b_tc3  = (const float*)d_in[14];

    const int B = in_sizes[0] / 7;

    _Float16* s2 = (_Float16*)d_ws;                              // (B,5,16,16) f16
    size_t s2_bytes = ((size_t)B * 1280 * sizeof(_Float16) + 255) & ~(size_t)255;
    _Float16* s4 = (_Float16*)((char*)d_ws + s2_bytes);          // (B,25,32,8) f16

    k1_lin_up1_tc1<<<B, 256, 0, stream>>>(latent, w_lin, b_lin, w_up1, b_up1,
                                          w_tc1, b_tc1, s2);
    k2_up2_tc2<<<B, 256, 0, stream>>>(s2, w_up2, b_up2, w_tc2, b_tc2, s4);
    k3_up3_tc3<<<B, 256, 0, stream>>>(s4, w_up3, b_up3, w_tc3, b_tc3, (float*)d_out);
}
